// CrissCrossAttention_90804198572459
// MI455X (gfx1250) — matrix operand order repaired and checked by reading, not compiled, not run on hardware
//
#include <hip/hip_runtime.h>

// CrissCross/self-attention for MI455X (gfx1250), wave32 + v_wmma_f32_16x16x32_f16.
// Compute-bound (~48 GFLOP vs ~70MB traffic) -> f16 WMMA with f32 accumulate,
// streaming (flash) softmax so the 268MB energy matrix never hits memory.
// Round 3: software pipelining — issue all independent loads of a chunk before
// its WMMA burst (partial loadcnt waits instead of full drains); next-chunk K
// prefetch in the attention j-loop overlaps with softmax VALU + LDS roundtrip.

#define B_    4
#define C_    256
#define CQK_  32
#define N_    4096            // H*W
#define LOG2E 1.4426950408889634f

typedef __attribute__((ext_vector_type(16))) _Float16 v16h;
typedef __attribute__((ext_vector_type(8)))  float    v8f;

union ABv { v16h v; uint4 u[2]; };

// A operand (16x32 f16, M rows = lane%16):
//   VGPR0-3: K = (hi?8..15:0..7), VGPR4-7: K = (hi?24..31:16..23)
__device__ __forceinline__ v16h load_a16(const _Float16* rowp, int hi) {
  ABv r;
  r.u[0] = *(const uint4*)(rowp + (hi ? 8 : 0));
  r.u[1] = *(const uint4*)(rowp + 16 + (hi ? 8 : 0));
  return r.v;
}
// B operand (32x16 f16, N col = lane%16):
//   VGPR0-3: K = (hi?8..15:0..7), VGPR4-7: K = (hi?24..31:16..23)
__device__ __forceinline__ v16h load_b16(const _Float16* colp, int hi) {
  ABv r;
  const _Float16* p = colp + (hi ? 8 : 0);
  r.u[0] = *(const uint4*)(p);
  r.u[1] = *(const uint4*)(p + 16);
  return r.v;
}
__device__ __forceinline__ v8f wmma16(v16h a, v16h b, v8f c) {
  return __builtin_amdgcn_wmma_f32_16x16x32_f16(false, a, false, b, (short)0, c, false, false);
}

#if defined(__AMDGCN__) && __has_builtin(__builtin_amdgcn_rcpf)
#define RCPF(x) __builtin_amdgcn_rcpf(x)
#else
#define RCPF(x) (1.0f / (x))
#endif

// ---------------- stage 0a: weights f32 -> f16 ----------------
__global__ void k_cvt_w(const float* __restrict__ wq, const float* __restrict__ wk,
                        const float* __restrict__ wv, const float* __restrict__ wg,
                        _Float16* __restrict__ wqh, _Float16* __restrict__ wkh,
                        _Float16* __restrict__ wvh, _Float16* __restrict__ wgh) {
  int i = blockIdx.x * blockDim.x + threadIdx.x;   // 0..65535
  if (i < CQK_ * C_) { wqh[i] = (_Float16)wq[i]; wkh[i] = (_Float16)wk[i]; }
  wvh[i] = (_Float16)wv[i];
  wgh[i] = (_Float16)wg[i];
}

// ---------------- stage 0b: x[b][c][n] f32 -> xh[b][n][c] f16 (tiled transpose) --------
__global__ void k_transpose_x(const float* __restrict__ x, _Float16* __restrict__ xh) {
  __shared__ float tile[32][33];
  const int b  = blockIdx.z;
  const int n0 = blockIdx.x * 32;
  const int c0 = blockIdx.y * 32;
  const int tx = threadIdx.x, ty = threadIdx.y;    // 32 x 8
  const float* xb = x + (size_t)b * C_ * N_;
#pragma unroll
  for (int k = 0; k < 32; k += 8)
    tile[ty + k][tx] = xb[(size_t)(c0 + ty + k) * N_ + n0 + tx];
  __syncthreads();
  _Float16* xo = xh + (size_t)b * N_ * C_;
#pragma unroll
  for (int k = 0; k < 32; k += 8)
    xo[(size_t)(n0 + ty + k) * C_ + c0 + tx] = (_Float16)tile[tx][ty + k];
}

// ---------------- stage 1a: Q/K projection (32 acc VGPRs / wave) -----------------------
__global__ void __launch_bounds__(256, 1)
k_qk(const _Float16* __restrict__ xh,
     const _Float16* __restrict__ wqh, const _Float16* __restrict__ wkh,
     const float* __restrict__ bq, const float* __restrict__ bk,
     _Float16* __restrict__ qh, _Float16* __restrict__ kh) {
  const int t    = blockIdx.x * 8 + (threadIdx.x >> 5);   // 0..1023
  const int b    = t >> 8;
  const int n0   = (t & 255) * 16;
  const int lane = threadIdx.x & 31;
  const int ln   = lane & 15;
  const int hi   = lane >> 4;

  v8f accq[2] = {}, acck[2] = {};
  const _Float16* xrow = xh + ((size_t)b * N_ + n0 + ln) * C_;
#pragma unroll
  for (int cc = 0; cc < 8; ++cc) {
    const int cb = cc * 32;
    v16h a = load_a16(xrow + cb, hi);
    v16h bmq[2], bmk[2];
#pragma unroll
    for (int dt = 0; dt < 2; ++dt) {       // all loads first -> partial waits
      bmq[dt] = load_b16(wqh + (size_t)(dt * 16 + ln) * C_ + cb, hi);
      bmk[dt] = load_b16(wkh + (size_t)(dt * 16 + ln) * C_ + cb, hi);
    }
#pragma unroll
    for (int dt = 0; dt < 2; ++dt) {
      accq[dt] = wmma16(a, bmq[dt], accq[dt]);
      acck[dt] = wmma16(a, bmk[dt], acck[dt]);
    }
  }
#pragma unroll
  for (int dt = 0; dt < 2; ++dt) {
    const int d = dt * 16 + ln;
    const float biq = bq[d], bik = bk[d];
#pragma unroll
    for (int p = 0; p < 8; ++p) {
      const size_t n = (size_t)b * N_ + n0 + p + hi * 8;
      qh[n * CQK_ + d] = (_Float16)((accq[dt][p] + biq) * LOG2E);  // fold log2e into Q
      kh[n * CQK_ + d] = (_Float16)(acck[dt][p] + bik);
    }
  }
}

// ---------------- stage 1b: V projection, half of C per wave (64 acc VGPRs) ------------
// v stored transposed: [b][c][n]
__global__ void __launch_bounds__(256, 1)
k_v(const _Float16* __restrict__ xh, const _Float16* __restrict__ wvh,
    const float* __restrict__ bv, _Float16* __restrict__ vh) {
  const int t    = blockIdx.x * 8 + (threadIdx.x >> 5);   // 0..2047
  const int b    = t >> 9;
  const int n0   = ((t >> 1) & 255) * 16;
  const int ch0  = (t & 1) * 8;                           // c-tile half
  const int lane = threadIdx.x & 31;
  const int ln   = lane & 15;
  const int hi   = lane >> 4;

  v8f accv[8] = {};
  const _Float16* xrow = xh + ((size_t)b * N_ + n0 + ln) * C_;
  const _Float16* wrow = wvh + (size_t)(ch0 * 16 + ln) * C_;
#pragma unroll
  for (int cc = 0; cc < 8; ++cc) {
    const int cb = cc * 32;
    v16h a = load_a16(xrow + cb, hi);
    v16h bm[8];
#pragma unroll
    for (int dt = 0; dt < 8; ++dt)         // all loads first -> partial waits
      bm[dt] = load_b16(wrow + (size_t)(dt * 16) * C_ + cb, hi);
#pragma unroll
    for (int dt = 0; dt < 8; ++dt)
      accv[dt] = wmma16(a, bm[dt], accv[dt]);
  }
#pragma unroll
  for (int dt = 0; dt < 8; ++dt) {
    const int c = (ch0 + dt) * 16 + ln;
    const float biv = bv[c];
#pragma unroll
    for (int p = 0; p < 8; ++p)
      vh[((size_t)b * C_ + c) * N_ + n0 + p + hi * 8] = (_Float16)(accv[dt][p] + biv);
  }
}

// ---------------- stage 2: streaming-softmax attention ---------------------------------
// wave -> 16 queries x 128 channels (8 acc tiles = 64 VGPRs); the paired wave
// recomputes S/softmax for the other 128 channels (QK is only ~12% of FLOPs).
__global__ void __launch_bounds__(256, 1)
k_attn(const _Float16* __restrict__ qh, const _Float16* __restrict__ kh,
       const _Float16* __restrict__ vh, _Float16* __restrict__ ao) {
  __shared__ __align__(16) _Float16 plds[8 * 16 * 40];    // 8 waves, 16x32 P, pitch 40h
  const int t    = blockIdx.x * 8 + (threadIdx.x >> 5);   // 0..2047
  const int b    = t >> 9;
  const int i0   = ((t >> 1) & 255) * 16;
  const int ch0  = (t & 1) * 8;                           // c-tile half
  const int lane = threadIdx.x & 31;
  const int ln   = lane & 15;
  const int hi   = lane >> 4;
  _Float16* pw   = plds + (threadIdx.x >> 5) * 640;

  const _Float16* khb = kh + (size_t)b * N_ * CQK_;
  const _Float16* vhb = vh + ((size_t)b * C_ + ch0 * 16 + ln) * N_;

  const v16h aq = load_a16(qh + ((size_t)b * N_ + i0 + ln) * CQK_, hi);

  v8f acc[8] = {};
  float rmax[8], lacc[8];
#pragma unroll
  for (int p = 0; p < 8; ++p) { rmax[p] = -INFINITY; lacc[p] = 0.0f; }
  const v8f zero = {};

  // prologue: K operands for chunk 0
  v16h bk0 = load_b16(khb + (size_t)ln * CQK_, hi);
  v16h bk1 = load_b16(khb + (size_t)(16 + ln) * CQK_, hi);

  for (int jc = 0; jc < N_ / 32; ++jc) {
    const int j0 = jc * 32;
    v8f s0 = wmma16(aq, bk0, zero);        // log2-domain logits (log2e folded in Q)
    v8f s1 = wmma16(aq, bk1, zero);

    // issue the chunk's 8 V operands early: they hide behind softmax + LDS work
    v16h bvv[8];
#pragma unroll
    for (int ct = 0; ct < 8; ++ct)
      bvv[ct] = load_b16(vhb + (size_t)(ct * 16) * N_ + j0, hi);

    // branchless next-chunk K prefetch (wraps to 0 on the last iteration)
    const int j1 = (jc + 1 < N_ / 32) ? j0 + 32 : 0;
    bk0 = load_b16(khb + (size_t)(j1 + ln) * CQK_, hi);
    bk1 = load_b16(khb + (size_t)(j1 + 16 + ln) * CQK_, hi);

    float sc[8];
#pragma unroll
    for (int p = 0; p < 8; ++p) {
      float m = fmaxf(s0[p], s1[p]);       // row lives in 16 lanes of one half
      m = fmaxf(m, __shfl_xor(m, 1));
      m = fmaxf(m, __shfl_xor(m, 2));
      m = fmaxf(m, __shfl_xor(m, 4));
      m = fmaxf(m, __shfl_xor(m, 8));
      const float nm    = fmaxf(rmax[p], m);
      const float scale = exp2f(rmax[p] - nm);
      rmax[p] = nm;
      const float e0 = exp2f(s0[p] - nm);
      const float e1 = exp2f(s1[p] - nm);
      lacc[p] = lacc[p] * scale + e0 + e1; // per-lane partial row sums
      sc[p]   = scale;
      const int r = p + hi * 8;            // C-layout -> A-layout via private LDS tile
      pw[r * 40 + ln]      = (_Float16)e0;
      pw[r * 40 + 16 + ln] = (_Float16)e1;
    }
#pragma unroll
    for (int ct = 0; ct < 8; ++ct)
#pragma unroll
      for (int p = 0; p < 8; ++p)
        acc[ct][p] *= sc[p];

    const v16h ap = load_a16(pw + ln * 40, hi);   // same-wave DS pipe is in-order
#pragma unroll
    for (int ct = 0; ct < 8; ++ct)
      acc[ct] = wmma16(ap, bvv[ct], acc[ct]);
  }

  float linv[8];
#pragma unroll
  for (int p = 0; p < 8; ++p) {
    float l = lacc[p];
    l += __shfl_xor(l, 1);
    l += __shfl_xor(l, 2);
    l += __shfl_xor(l, 4);
    l += __shfl_xor(l, 8);
    linv[p] = RCPF(l);
  }
  _Float16* aob = ao + (size_t)b * N_ * C_;
#pragma unroll
  for (int ct = 0; ct < 8; ++ct) {
    const int c = (ch0 + ct) * 16 + ln;
#pragma unroll
    for (int p = 0; p < 8; ++p)
      aob[(size_t)(i0 + p + hi * 8) * C_ + c] = (_Float16)(acc[ct][p] * linv[p]);
  }
}

// ---------------- stage 3: output projection + bias + residual -------------------------
__global__ void __launch_bounds__(256, 1)
k_proj(const float* __restrict__ x, const _Float16* __restrict__ wgh,
       const float* __restrict__ bg, const _Float16* __restrict__ ao,
       float* __restrict__ out) {
  const int t    = blockIdx.x * 8 + (threadIdx.x >> 5);   // 0..4095
  const int b    = t >> 10;
  const int o0   = ((t >> 6) & 15) * 16;
  const int n0   = (t & 63) * 64;
  const int lane = threadIdx.x & 31;
  const int ln   = lane & 15;
  const int hi   = lane >> 4;

  v8f acc[4] = {};
  const _Float16* wrow = wgh + (size_t)(o0 + ln) * C_;
  const _Float16* aob  = ao + (size_t)b * N_ * C_;
#pragma unroll
  for (int cc = 0; cc < 8; ++cc) {
    const int cb = cc * 32;
    v16h a = load_a16(wrow + cb, hi);
    v16h bm[4];
#pragma unroll
    for (int st = 0; st < 4; ++st)         // all loads first -> partial waits
      bm[st] = load_b16(aob + (size_t)(n0 + st * 16 + ln) * C_ + cb, hi);
#pragma unroll
    for (int st = 0; st < 4; ++st)
      acc[st] = wmma16(a, bm[st], acc[st]);
  }
#pragma unroll
  for (int p = 0; p < 8; ++p) {
    const int o = o0 + p + hi * 8;
    const float bias = bg[o];
    const size_t base = ((size_t)b * C_ + o) * N_ + n0 + ln;
#pragma unroll
    for (int st = 0; st < 4; ++st) {
      const size_t idx = base + st * 16;
      out[idx] = x[idx] + acc[st][p] + bias;
    }
  }
}

// ---------------- launch ----------------------------------------------------------------
extern "C" void kernel_launch(void* const* d_in, const int* in_sizes, int n_in,
                              void* d_out, int out_size, void* d_ws, size_t ws_size,
                              hipStream_t stream) {
  (void)in_sizes; (void)n_in; (void)out_size; (void)ws_size;
  const float* x  = (const float*)d_in[0];
  const float* wq = (const float*)d_in[1];
  const float* bq = (const float*)d_in[2];
  const float* wk = (const float*)d_in[3];
  const float* bk = (const float*)d_in[4];
  const float* wv = (const float*)d_in[5];
  const float* bv = (const float*)d_in[6];
  const float* wg = (const float*)d_in[7];
  const float* bg = (const float*)d_in[8];

  char* w = (char*)d_ws;                                  // ~26.2 MiB total
  _Float16* xh  = (_Float16*)(w);                         // [B][N][C]  8 MiB
  _Float16* vh  = (_Float16*)(w + (8u  << 20));           // [B][C][N]  8 MiB
  _Float16* qh  = (_Float16*)(w + (16u << 20));           // [B][N][32] 1 MiB
  _Float16* kh  = (_Float16*)(w + (17u << 20));           // [B][N][32] 1 MiB
  _Float16* ao  = (_Float16*)(w + (18u << 20));           // [B][N][C]  8 MiB
  _Float16* wqh = (_Float16*)(w + (26u << 20));
  _Float16* wkh = wqh + CQK_ * C_;
  _Float16* wvh = wkh + CQK_ * C_;
  _Float16* wgh = wvh + C_ * C_;

  k_cvt_w<<<C_ * C_ / 256, 256, 0, stream>>>(wq, wk, wv, wg, wqh, wkh, wvh, wgh);
  dim3 gt(N_ / 32, C_ / 32, B_), bt(32, 8);
  k_transpose_x<<<gt, bt, 0, stream>>>(x, xh);
  k_qk  <<<128, 256, 0, stream>>>(xh, wqh, wkh, bq, bk, qh, kh);
  k_v   <<<256, 256, 0, stream>>>(xh, wvh, bv, vh);
  k_attn<<<256, 256, 0, stream>>>(qh, kh, vh, ao);
  k_proj<<<512, 256, 0, stream>>>(x, wgh, bg, ao, (float*)d_out);
}
